// GNN_28389733826682
// MI455X (gfx1250) — matrix operand order repaired and checked by reading, not compiled, not run on hardware
//
#include <hip/hip_runtime.h>
#include <math.h>

// ---------------------------------------------------------------------------
// GCN 2-layer inference for MI455X (gfx1250, wave32, WMMA).
// GEMMs in bf16 via v_wmma_f32_16x16x32_bf16 with double-buffered
// GLOBAL_LOAD_ASYNC_TO_LDS_B128 tile staging (copy/compute overlap via
// ASYNCcnt); aggregation via fp32 atomics with uniform index loads;
// pre-aggregation activations stored bf16 to halve gather traffic.
// ---------------------------------------------------------------------------

typedef __bf16 bf16_t;
typedef bf16_t v16bf __attribute__((ext_vector_type(16)));
typedef float  v8f   __attribute__((ext_vector_type(8)));
typedef unsigned int   u32x4 __attribute__((ext_vector_type(4)));

union Frag {            // 32 bytes: one WMMA A or B operand (wave32 layout)
  v16bf v;
  u32x4 q[2];
};

__device__ __forceinline__ unsigned short f2bf(float f) {
  unsigned int u = __float_as_uint(f);
  u += 0x7FFFu + ((u >> 16) & 1u);           // round-to-nearest-even
  return (unsigned short)(u >> 16);
}
__device__ __forceinline__ float bf2f(unsigned short h) {
  return __uint_as_float(((unsigned int)h) << 16);
}

// Async global->LDS 16B copy (per lane). VDST = LDS byte address, VADDR = VA.
// Generic pointers to __shared__ carry the LDS offset in their low 32 bits.
__device__ __forceinline__ void async_copy_b128(unsigned lds_addr, const void* gaddr) {
  asm volatile("global_load_async_to_lds_b128 %0, %1, off"
               :: "v"(lds_addr), "v"(gaddr) : "memory");
}
__device__ __forceinline__ void wait_async0() {
  asm volatile("s_wait_asynccnt 0x0" ::: "memory");
}
__device__ __forceinline__ void wait_async3() {   // first-in-order ops complete
  asm volatile("s_wait_asynccnt 0x3" ::: "memory");
}

// ---------------------------- elementwise helpers --------------------------

__global__ void gcn_fill_f32(float* p, float v, long n) {
  long i = (long)blockIdx.x * blockDim.x + threadIdx.x;
  if (i < n) p[i] = v;
}

__global__ void gcn_cvt_f32_bf16(const float* __restrict__ in,
                                 unsigned short* __restrict__ out, long n) {
  long i = (long)blockIdx.x * blockDim.x + threadIdx.x;
  if (i < n) out[i] = f2bf(in[i]);
}

// Wt[n*K + k] = bf16(W[k*N + n]), zero-padded for n >= N (Npad columns total)
__global__ void gcn_w_transpose_bf16(const float* __restrict__ W,
                                     unsigned short* __restrict__ Wt,
                                     int K, int N, int Npad) {
  long i = (long)blockIdx.x * blockDim.x + threadIdx.x;
  if (i >= (long)Npad * K) return;
  int n = (int)(i / K);
  int k = (int)(i - (long)n * K);
  Wt[i] = (n < N) ? f2bf(W[(long)k * N + n]) : (unsigned short)0;
}

__global__ void gcn_deg_accum(const int* __restrict__ dst, float* deg, int E) {
  int e = blockIdx.x * blockDim.x + threadIdx.x;
  if (e < E) atomicAdd(&deg[dst[e]], 1.0f);
}

__global__ void gcn_rsqrt_inplace(float* p, int n) {
  int i = blockIdx.x * blockDim.x + threadIdx.x;
  if (i < n) p[i] = rsqrtf(p[i]);           // deg >= 1 (self-loop), no zero case
}

__global__ void gcn_edge_norm(const int* __restrict__ src, const int* __restrict__ dst,
                              const float* __restrict__ dinv, float* __restrict__ nrm, int E) {
  int e = blockIdx.x * blockDim.x + threadIdx.x;
  if (e < E) nrm[e] = dinv[src[e]] * dinv[dst[e]];
}

// --------------------------- WMMA tiled GEMM -------------------------------
// C[M,N] = A[M,K] * Bt^T where Bt is [Npad][K] (pre-transposed, zero-padded
// weights). Block tile 128x64, 8 waves each computing 32x32 (2x2 WMMA
// 16x16x32 tiles). K staged in 32-chunks, double-buffered in LDS with async
// global->LDS b128 copies overlapping the WMMAs of the previous chunk.
template <bool OUT_BF16>
__global__ __launch_bounds__(256) void gcn_gemm_wmma(
    const unsigned short* __restrict__ A, const unsigned short* __restrict__ Bt,
    void* __restrict__ C, int M, int N, int K) {
  // strides of 48 ushorts (96B) keep every 16B LDS access aligned
  __shared__ unsigned short As[2][128 * 48];   // As[p][m][k]  (k = 0..31 of chunk)
  __shared__ unsigned short Bs[2][64 * 48];    // Bs[p][n][k]

  const int tid  = threadIdx.x;
  const int m0   = blockIdx.x * 128;
  const int n0   = blockIdx.y * 64;
  const int wave = tid >> 5;
  const int lane = tid & 31;
  const int wm   = (wave & 3) << 5;         // wave M offset in block tile
  const int wn   = (wave >> 2) << 5;        // wave N offset in block tile

  // A staging: 2 x (row, col8); rows clamped so tail blocks stay branch-free
  // (clamped rows only feed C rows that the epilogue never stores).
  const int ar0 = tid >> 2;                 // 0..63
  const int ar1 = (tid + 256) >> 2;         // 64..127
  const int ac  = (tid & 3) << 3;           // 0,8,16,24
  const long amax = (long)(M - 1);
  const long ga0 = ((long)(m0 + ar0) > amax ? amax : (long)(m0 + ar0)) * K + ac;
  const long ga1 = ((long)(m0 + ar1) > amax ? amax : (long)(m0 + ar1)) * K + ac;
  // B staging: 1 x (n, col8); Bt is padded to gridDim.y*64 columns
  const int bn = tid >> 2;                  // 0..63
  const long gb = (long)(n0 + bn) * K + ac;

  const unsigned ABUF = 128 * 48 * 2;       // bytes per LDS buffer
  const unsigned BBUF = 64 * 48 * 2;
  const unsigned lA0 = (unsigned)(size_t)&As[0][ar0 * 48 + ac];
  const unsigned lA1 = (unsigned)(size_t)&As[0][ar1 * 48 + ac];
  const unsigned lB  = (unsigned)(size_t)&Bs[0][bn * 48 + ac];

  v8f acc[2][2];
  #pragma unroll
  for (int i = 0; i < 2; ++i)
    #pragma unroll
    for (int j = 0; j < 2; ++j)
      #pragma unroll
      for (int r = 0; r < 8; ++r) acc[i][j][r] = 0.0f;

  // fragment coordinates (ISA 7.12.2 wave32 layouts)
  const int fm  = lane & 15;
  const int kbA = (lane >> 4) << 3;         // A: 0 or 8
  const int kbB = (lane >> 4) << 4;         // B: 0 or 16

  const int nch = K >> 5;
  // prologue: chunk 0 into buffer 0
  async_copy_b128(lA0, A + ga0);
  async_copy_b128(lA1, A + ga1);
  async_copy_b128(lB, Bt + gb);

  for (int i = 0; i < nch; ++i) {
    const int p = i & 1;
    if (i + 1 < nch) {                      // prefetch next chunk into other buffer
      const int q = p ^ 1;
      const int kk = (i + 1) << 5;
      async_copy_b128(lA0 + q * ABUF, A + ga0 + kk);
      async_copy_b128(lA1 + q * ABUF, A + ga1 + kk);
      async_copy_b128(lB + q * BBUF, Bt + gb + kk);
      wait_async3();                        // in-order: chunk i's 3 copies done
    } else {
      wait_async0();
    }
    __syncthreads();

    const unsigned short* __restrict__ ab = &As[p][0];
    const unsigned short* __restrict__ bb = &Bs[p][0];
    Frag a0, a1, b0, b1;
    // A 16x32 bf16: lanes 0-15 -> M=lane, K {kb..kb+7, 16+kb..}; kb = 8*(lane>>4)
    a0.q[0] = *(const u32x4*)&ab[(wm + fm) * 48 + kbA];
    a0.q[1] = *(const u32x4*)&ab[(wm + fm) * 48 + 16 + kbA];
    a1.q[0] = *(const u32x4*)&ab[(wm + 16 + fm) * 48 + kbA];
    a1.q[1] = *(const u32x4*)&ab[(wm + 16 + fm) * 48 + 16 + kbA];
    // B 32x16 bf16: lanes 0-15 -> N=lane, K {kb..kb+7, 16+kb..}; kb = 8*(lane>>4)
    b0.q[0] = *(const u32x4*)&bb[(wn + fm) * 48 + kbA];
    b0.q[1] = *(const u32x4*)&bb[(wn + fm) * 48 + 16 + kbA];
    b1.q[0] = *(const u32x4*)&bb[(wn + 16 + fm) * 48 + kbA];
    b1.q[1] = *(const u32x4*)&bb[(wn + 16 + fm) * 48 + 16 + kbA];

    acc[0][0] = __builtin_amdgcn_wmma_f32_16x16x32_bf16(false, a0.v, false, b0.v,
                                                        (short)0, acc[0][0], false, false);
    acc[0][1] = __builtin_amdgcn_wmma_f32_16x16x32_bf16(false, a0.v, false, b1.v,
                                                        (short)0, acc[0][1], false, false);
    acc[1][0] = __builtin_amdgcn_wmma_f32_16x16x32_bf16(false, a1.v, false, b0.v,
                                                        (short)0, acc[1][0], false, false);
    acc[1][1] = __builtin_amdgcn_wmma_f32_16x16x32_bf16(false, a1.v, false, b1.v,
                                                        (short)0, acc[1][1], false, false);
    __syncthreads();
  }

  // --- epilogue: D layout = VGPR r, lanes 0-15 M=r, lanes 16-31 M=8+r ---
  const int colL = lane & 15;
  const int rowS = (lane >> 4) << 3;        // 0 or 8
  if (m0 + 128 <= M && n0 + 64 <= N) {
    // full tile: unguarded stores, 32-bit offsets (fits: M*N < 2^31)
    #pragma unroll
    for (int i = 0; i < 2; ++i)
      #pragma unroll
      for (int j = 0; j < 2; ++j) {
        unsigned base = (unsigned)((m0 + wm + (i << 4) + rowS) * N
                                   + n0 + wn + (j << 4) + colL);
        #pragma unroll
        for (int r = 0; r < 8; ++r) {
          float v = acc[i][j][r];
          if (OUT_BF16) ((unsigned short*)C)[base + (unsigned)(r * N)] = f2bf(v);
          else          ((float*)C)[base + (unsigned)(r * N)] = v;
        }
      }
  } else {
    #pragma unroll
    for (int i = 0; i < 2; ++i)
      #pragma unroll
      for (int j = 0; j < 2; ++j)
        #pragma unroll
        for (int r = 0; r < 8; ++r) {
          int gm = m0 + wm + (i << 4) + rowS + r;
          int gn = n0 + wn + (j << 4) + colL;
          if (gm < M && gn < N) {
            float v = acc[i][j][r];
            unsigned off = (unsigned)(gm * N + gn);
            if (OUT_BF16) ((unsigned short*)C)[off] = f2bf(v);
            else          ((float*)C)[off] = v;
          }
        }
  }
}

// --------------------------- edge aggregation ------------------------------

// one block per edge: src/dst/norm become uniform (scalar) loads
__global__ void gcn_agg1(const unsigned short* __restrict__ hb,
                         const int* __restrict__ src, const int* __restrict__ dst,
                         const float* __restrict__ nrm, float* __restrict__ agg) {
  const int e = blockIdx.x;
  const int c = threadIdx.x;                // channels c and c+256
  const int s = src[e], d = dst[e];
  const float w = nrm[e];
  const long sb = (long)s * 512, db = (long)d * 512;
  atomicAdd(&agg[db + c],       bf2f(hb[sb + c])       * w);
  atomicAdd(&agg[db + c + 256], bf2f(hb[sb + c + 256]) * w);
}

// relu(agg + dinv^2 * h_self + b1) -> bf16 (in place over hb)
__global__ void gcn_post1(const float* __restrict__ agg,
                          const unsigned short* __restrict__ hb,
                          const float* __restrict__ dinv, const float* __restrict__ b1,
                          unsigned short* __restrict__ out, long total) {
  long idx = (long)blockIdx.x * blockDim.x + threadIdx.x;
  if (idx >= total) return;
  int i = (int)(idx >> 9);
  int c = (int)(idx & 511);
  float d = dinv[i];
  float v = agg[idx] + d * d * bf2f(hb[idx]) + b1[c];
  out[idx] = f2bf(fmaxf(v, 0.0f));
}

// one wave32 per edge; lane covers class c and c+32
__global__ void gcn_agg2(const float* __restrict__ h2,
                         const int* __restrict__ src, const int* __restrict__ dst,
                         const float* __restrict__ nrm, float* __restrict__ agg, int E) {
  int e    = (blockIdx.x * blockDim.x + threadIdx.x) >> 5;
  int lane = threadIdx.x & 31;
  if (e >= E) return;
  const int s = src[e], d = dst[e];
  const float w = nrm[e];
  const long sb = (long)s * 40, db = (long)d * 40;
  atomicAdd(&agg[db + lane], h2[sb + lane] * w);
  if (lane < 8) atomicAdd(&agg[db + 32 + lane], h2[sb + 32 + lane] * w);
}

// self-loop + bias + log_softmax; one wave32 per node, lane covers c and c+32
__global__ void gcn_final(const float* __restrict__ agg, const float* __restrict__ h2,
                          const float* __restrict__ dinv, const float* __restrict__ b2,
                          float* __restrict__ out, int Nn) {
  int node = blockIdx.x * (blockDim.x >> 5) + (threadIdx.x >> 5);
  int lane = threadIdx.x & 31;
  if (node >= Nn) return;
  float d  = dinv[node];
  float d2 = d * d;
  long base = (long)node * 40;
  int c0 = lane, c1 = lane + 32;
  float l0 = agg[base + c0] + d2 * h2[base + c0] + b2[c0];     // c0 < 40 always
  float l1 = -1e30f;
  if (c1 < 40) l1 = agg[base + c1] + d2 * h2[base + c1] + b2[c1];
  float m = fmaxf(l0, l1);
  #pragma unroll
  for (int off = 16; off >= 1; off >>= 1) m = fmaxf(m, __shfl_xor(m, off, 32));
  float s = expf(l0 - m);
  if (c1 < 40) s += expf(l1 - m);
  #pragma unroll
  for (int off = 16; off >= 1; off >>= 1) s += __shfl_xor(s, off, 32);
  float lse = m + logf(s);
  out[base + c0] = l0 - lse;
  if (c1 < 40) out[base + c1] = l1 - lse;
}

// ------------------------------- launcher ----------------------------------

extern "C" void kernel_launch(void* const* d_in, const int* in_sizes, int n_in,
                              void* d_out, int out_size, void* d_ws, size_t ws_size,
                              hipStream_t stream) {
  const float* x  = (const float*)d_in[0];
  const int*   ei = (const int*)d_in[1];
  const float* W1 = (const float*)d_in[2];
  const float* b1 = (const float*)d_in[3];
  const float* W2 = (const float*)d_in[4];
  const float* b2 = (const float*)d_in[5];

  const int Nn  = in_sizes[0] / 128;   // 100000
  const int E   = in_sizes[1] / 2;     // 500000
  const int IN  = 128, HID = 512, NC = 40;
  const int NC_PAD = 64;               // GEMM2 N tile width (zero-padded weights)
  const int* src = ei;
  const int* dst = ei + E;

  // carve workspace (256B-aligned slices)
  char* ws = (char*)d_ws;
  auto carve = [&](size_t bytes) {
    char* p = ws;
    ws += (bytes + 255) & ~(size_t)255;
    return p;
  };
  float*          deg  = (float*)carve((size_t)Nn * 4);          // becomes dinv
  float*          nrm  = (float*)carve((size_t)E * 4);
  unsigned short* xb   = (unsigned short*)carve((size_t)Nn * IN * 2);
  unsigned short* w1t  = (unsigned short*)carve((size_t)HID * IN * 2);     // [512][128]
  unsigned short* w2t  = (unsigned short*)carve((size_t)NC_PAD * HID * 2); // [64][512]
  unsigned short* hb   = (unsigned short*)carve((size_t)Nn * HID * 2);     // h1 pre-agg -> relu'd h1
  float*          agg1 = (float*)carve((size_t)Nn * HID * 4);
  float*          h2   = (float*)carve((size_t)Nn * NC * 4);
  float*          agg2 = (float*)carve((size_t)Nn * NC * 4);
  (void)ws_size;

  const int T = 256;
  auto blocks = [&](long n) { return (unsigned)((n + T - 1) / T); };

  // 1) degrees (self-loop contributes 1) -> dinv -> per-edge norm
  gcn_fill_f32<<<blocks(Nn), T, 0, stream>>>(deg, 1.0f, Nn);
  gcn_deg_accum<<<blocks(E), T, 0, stream>>>(dst, deg, E);
  gcn_rsqrt_inplace<<<blocks(Nn), T, 0, stream>>>(deg, Nn);
  gcn_edge_norm<<<blocks(E), T, 0, stream>>>(src, dst, deg, nrm, E);

  // 2) bf16 conversions; weights pre-transposed+padded into [Npad][K]
  gcn_cvt_f32_bf16<<<blocks((long)Nn * IN), T, 0, stream>>>(x, xb, (long)Nn * IN);
  gcn_w_transpose_bf16<<<blocks((long)HID * IN), T, 0, stream>>>(W1, w1t, IN, HID, HID);
  gcn_w_transpose_bf16<<<blocks((long)NC_PAD * HID), T, 0, stream>>>(W2, w2t, HID, NC, NC_PAD);

  // 3) GEMM1: hb = xb @ W1   [Nn,128]x[128,512] -> bf16
  {
    dim3 grid((Nn + 127) / 128, HID / 64);
    gcn_gemm_wmma<true><<<grid, T, 0, stream>>>(xb, w1t, (void*)hb, Nn, HID, IN);
  }

  // 4) layer-1 aggregation (scatter atomics) + self-loop/bias/relu -> bf16 in place
  gcn_fill_f32<<<blocks((long)Nn * HID), T, 0, stream>>>(agg1, 0.0f, (long)Nn * HID);
  gcn_agg1<<<E, T, 0, stream>>>(hb, src, dst, nrm, agg1);
  gcn_post1<<<blocks((long)Nn * HID), T, 0, stream>>>(agg1, hb, deg, b1, hb, (long)Nn * HID);

  // 5) GEMM2: h2 = hb @ W2   [Nn,512]x[512,40] -> f32 (N padded to 64-wide tile)
  {
    dim3 grid((Nn + 127) / 128, 1);
    gcn_gemm_wmma<false><<<grid, T, 0, stream>>>(hb, w2t, (void*)h2, Nn, NC, HID);
  }

  // 6) layer-2 aggregation + fused self-loop/bias/log-softmax
  gcn_fill_f32<<<blocks((long)Nn * NC), T, 0, stream>>>(agg2, 0.0f, (long)Nn * NC);
  gcn_agg2<<<blocks((long)E * 32), T, 0, stream>>>(h2, src, dst, nrm, agg2, E);
  gcn_final<<<(Nn + 3) / 4, 128, 0, stream>>>(agg2, h2, deg, b2, (float*)d_out, Nn);
}
